// MyModel_64819646431678
// MI455X (gfx1250) — compile-verified, matrix operand order checked by reading, not run on hardware
//
#include <hip/hip_runtime.h>
#include <hip/hip_bf16.h>

// ---------------- model constants ----------------
#define T_   2048
#define H_   1024
#define V_   32000
#define I_   2816
#define NH_  16
#define HD_  64
#define E_   8
#define L_   4
#define CAP_ 5120   // T_*K(=2) + E_*127 padding, 128-aligned expert segments

typedef __attribute__((ext_vector_type(16))) __bf16 v16bf;
typedef __attribute__((ext_vector_type(8)))  float  v8f;
typedef __attribute__((ext_vector_type(4)))  int    v4i_;

// ---------------- async global->LDS (CDNA5), guarded ----------------
#if defined(__has_builtin)
#if __has_builtin(__builtin_amdgcn_global_load_async_to_lds_b128)
#define HAVE_ASYNC_LDS 1
#endif
#endif

__device__ __forceinline__ void cp16_g2l(const __bf16* g, __bf16* l) {
#if defined(HAVE_ASYNC_LDS)
  __builtin_amdgcn_global_load_async_to_lds_b128((v4i_*)(void*)g, (v4i_*)(void*)l, 0, 0);
#else
  *(uint4*)l = *(const uint4*)g;
#endif
}
__device__ __forceinline__ void cp_wait() {
#if defined(HAVE_ASYNC_LDS)
#if __has_builtin(__builtin_amdgcn_s_wait_asynccnt)
  __builtin_amdgcn_s_wait_asynccnt(0);
#else
  asm volatile("s_wait_asynccnt 0x0" ::: "memory");
#endif
#endif
}

// ---------------- bf16 helpers ----------------
__device__ __forceinline__ __bf16 f2bf(float f) {
  unsigned u = __builtin_bit_cast(unsigned, f);
  u += 0x7fffu + ((u >> 16) & 1u);               // round-to-nearest-even
  unsigned short s = (unsigned short)(u >> 16);
  return __builtin_bit_cast(__bf16, s);
}
__device__ __forceinline__ float bf2f(__bf16 h) {
  unsigned short s = __builtin_bit_cast(unsigned short, h);
  unsigned u = ((unsigned)s) << 16;
  return __builtin_bit_cast(float, u);
}

// ---------------- WMMA fragment loads (contiguous b128 pairs) ----------------
struct uint4x2 { uint4 a, b; };
// 16 contiguous bf16 (pre-swizzled B layout: [group][lane][j])
__device__ __forceinline__ v16bf ld_frag_contig(const __bf16* p) {
  uint4x2 t;
  t.a = *(const uint4*)p;
  t.b = *(const uint4*)(p + 8);
  return __builtin_bit_cast(v16bf, t);
}
// A fragment: row-major tile; lane holds row m=lane&15; j0..7 at k=8*half+j,
// j8..15 at k=16+8*half+(j-8) -> two 8-elem runs 16 elems apart
__device__ __forceinline__ v16bf ld_frag_a(const __bf16* tile, int ldw, int mb, int kb, int lane) {
  const int half = lane >> 4;
  const __bf16* p = tile + (mb + (lane & 15)) * ldw + kb + 8 * half;
  uint4x2 t;
  t.a = *(const uint4*)p;
  t.b = *(const uint4*)(p + 16);
  return __builtin_bit_cast(v16bf, t);
}
__device__ __forceinline__ v8f wmma_bf16(v16bf a, v16bf b, v8f c) {
  return __builtin_amdgcn_wmma_f32_16x16x32_bf16(false, a, false, b, (short)0, c, false, false);
}

// ---------------- weight pre-pack: fp32 -> bf16 in fragment order ----------------
// Packed layout: tiles of 32k x 128n, tile = kt*(N/128)+nt, inside: g*512 + lane*16 + j
// with n = nt*128 + g*16 + (lane&15); k = kt*32 + 16*(j>>3) + 8*(lane>>4) + (j&7)
__global__ __launch_bounds__(256) void pack_w_kernel(
    const float* __restrict__ src, __bf16* __restrict__ dst,
    int Kd, int N, int transB, int nmat) {
  const long per = (long)Kd * N;
  const long total = per * nmat;
  long i = (long)blockIdx.x * 256 + threadIdx.x;
  if (i >= total) return;
  const int mat = (int)(i / per);
  const long r = i - (long)mat * per;
  const int ntn = N >> 7;
  const long tile = r >> 12;
  const int rem = (int)(r & 4095);
  const int kt = (int)(tile / ntn), nt = (int)(tile % ntn);
  const int g = rem >> 9, lane = (rem >> 4) & 31, j = rem & 15;
  const int half = lane >> 4;
  const int n = nt * 128 + g * 16 + (lane & 15);
  const int k = kt * 32 + ((j >> 3) << 4) + (half << 3) + (j & 7);
  const float* s = src + (long)mat * per;
  const float vv = transB ? s[(long)n * Kd + k] : s[(long)k * N + n];
  dst[i] = f2bf(vv);
}

// ---------------- generic bf16 GEMM (packed B, async double-buffered) ----------------
#define GBM 128
#define GBN 128
#define GBK 32
#define LDA_ 40      // 32 + 8 pad
#define BTILE 4096   // 32*128 elems

__global__ __launch_bounds__(256) void gemm_bf16_kernel(
    const __bf16* __restrict__ A, int lda,
    const __bf16* __restrict__ Bp, long strideB, int ntn,
    float* __restrict__ C, int ldc,
    int M, int N, int Kd,
    const float* __restrict__ residual,
    const int* __restrict__ rowIdx,
    const int* __restrict__ aoff) {
  __shared__ __bf16 As[2][GBM * LDA_];
  __shared__ __bf16 Bs[2][BTILE];

  const int tid = threadIdx.x, lane = tid & 31, w = tid >> 5;
  const int wm = (w >> 1) * 32, wn = (w & 1) * 64;
  const int m0 = blockIdx.x * GBM, n0 = blockIdx.y * GBN, nt = blockIdx.y;

  const __bf16* B = Bp;
  if (aoff) {
    int e = 0;
#pragma unroll
    for (int i = 0; i < E_; ++i)
      if (m0 >= aoff[i] && m0 < aoff[i + 1]) e = i;
    B = Bp + (long)e * strideB;
  }

  // A staging: 128 rows x 32 cols, thread = (row, 16-col half)
  const int arow = tid >> 1, acg = (tid & 1) * 16;
  int aIdx = -1;
  if (m0 + arow < M) aIdx = rowIdx ? rowIdx[m0 + arow] : (m0 + arow);
  if (aIdx < 0) {  // zero invalid rows once (both buffers; never overwritten)
    const uint4 z = make_uint4(0u, 0u, 0u, 0u);
    *(uint4*)&As[0][arow * LDA_ + acg] = z;
    *(uint4*)&As[0][arow * LDA_ + acg + 8] = z;
    *(uint4*)&As[1][arow * LDA_ + acg] = z;
    *(uint4*)&As[1][arow * LDA_ + acg + 8] = z;
  }

  v8f acc[2][4] = {};
  const int nkt = Kd / GBK;

  auto stage = [&](int buf, int kt) {
    if (aIdx >= 0) {
      const __bf16* ga = A + (long)aIdx * lda + kt * GBK + acg;
      cp16_g2l(ga, &As[buf][arow * LDA_ + acg]);
      cp16_g2l(ga + 8, &As[buf][arow * LDA_ + acg + 8]);
    }
    const __bf16* gb = B + ((long)kt * ntn + nt) * BTILE + tid * 16;
    cp16_g2l(gb, &Bs[buf][tid * 16]);
    cp16_g2l(gb + 8, &Bs[buf][tid * 16 + 8]);
  };

  stage(0, 0);
  for (int kt = 0; kt < nkt; ++kt) {
    cp_wait();
    __syncthreads();                         // tile kt ready; prev reads retired
    if (kt + 1 < nkt) stage((kt + 1) & 1, kt + 1);
    const int buf = kt & 1;
    const __bf16* Ab = &As[buf][0];
    const __bf16* Bb = &Bs[buf][0];
    v16bf af[2], bfm[4];
#pragma unroll
    for (int i = 0; i < 2; ++i) af[i] = ld_frag_a(Ab, LDA_, wm + i * 16, 0, lane);
#pragma unroll
    for (int f = 0; f < 4; ++f) bfm[f] = ld_frag_contig(Bb + ((wn >> 4) + f) * 512 + lane * 16);
#pragma unroll
    for (int i = 0; i < 2; ++i)
#pragma unroll
      for (int f = 0; f < 4; ++f)
        acc[i][f] = wmma_bf16(af[i], bfm[f], acc[i][f]);
  }

  // epilogue (C layout: m = elem + 8*half, n = lane&15)
  const int half = lane >> 4, nn = lane & 15;
#pragma unroll
  for (int i = 0; i < 2; ++i)
#pragma unroll
    for (int f = 0; f < 4; ++f) {
      const int n = n0 + wn + f * 16 + nn;
#pragma unroll
      for (int p = 0; p < 8; ++p) {
        const int m = m0 + wm + i * 16 + 8 * half + p;
        if (m < M && n < N) {
          float vv = acc[i][f][p];
          if (residual) vv += residual[(long)m * ldc + n];
          C[(long)m * ldc + n] = vv;
        }
      }
    }
}

// ---------------- embedding gather ----------------
__global__ __launch_bounds__(256) void gather_embed_kernel(
    const int* __restrict__ ids, const float* __restrict__ embed, float* __restrict__ h) {
  long i = (long)blockIdx.x * 256 + threadIdx.x;   // T_*H_
  int t = (int)(i >> 10), c = (int)(i & 1023);
  h[i] = embed[(long)ids[t] * H_ + c];
}

// ---------------- RMSNorm -> bf16 ----------------
__global__ __launch_bounds__(256) void rmsnorm_kernel(
    const float* __restrict__ x, const float* __restrict__ w, __bf16* __restrict__ out) {
  const int t = blockIdx.x, tid = threadIdx.x;
  const float* xr = x + (long)t * H_;
  float4 xv = *(const float4*)(xr + tid * 4);
  float ss = xv.x * xv.x + xv.y * xv.y + xv.z * xv.z + xv.w * xv.w;
  __shared__ float red[256];
  red[tid] = ss;
  __syncthreads();
  for (int s = 128; s > 0; s >>= 1) {
    if (tid < s) red[tid] += red[tid + s];
    __syncthreads();
  }
  const float inv = rsqrtf(red[0] / (float)H_ + 1e-6f);
  float4 wv = *(const float4*)(w + tid * 4);
  __bf16* o = out + (long)t * H_ + tid * 4;
  o[0] = f2bf(xv.x * inv * wv.x);
  o[1] = f2bf(xv.y * inv * wv.y);
  o[2] = f2bf(xv.z * inv * wv.z);
  o[3] = f2bf(xv.w * inv * wv.w);
}

// ---------------- RoPE: fp32 in -> rotated bf16 out ----------------
__global__ __launch_bounds__(256) void rope_bf16_kernel(
    const float* __restrict__ x, __bf16* __restrict__ out) {
  int idx = blockIdx.x * 256 + threadIdx.x;      // T_*NH_*32
  int d = idx & 31, hh = (idx >> 5) & 15, t = idx >> 9;
  float inv = __powf(10000.f, -((float)(2 * d)) / 64.f);
  float ang = (float)t * inv;
  float c = __cosf(ang), s = __sinf(ang);
  long base = (long)t * H_ + hh * HD_;
  float x1 = x[base + d], x2 = x[base + 32 + d];
  out[base + d]      = f2bf(x1 * c - x2 * s);
  out[base + 32 + d] = f2bf(x2 * c + x1 * s);
}

// ---------------- V transpose: fp32 [T,H] -> bf16 [H,T] ----------------
__global__ __launch_bounds__(256) void vtrans_kernel(
    const float* __restrict__ v, __bf16* __restrict__ vt) {
  long i = (long)blockIdx.x * 256 + threadIdx.x;   // H_*T_
  int h = (int)(i >> 11), t = (int)(i & 2047);
  vt[i] = f2bf(v[(long)t * H_ + h]);
}

// ---------------- flash attention (WMMA, causal, pre-swizzled K/V) ----------------
#define FLD 72
__global__ __launch_bounds__(128) void flash_attn_kernel(
    const __bf16* __restrict__ qb, const __bf16* __restrict__ kb,
    const __bf16* __restrict__ vt, __bf16* __restrict__ o) {
  __shared__ __bf16 Qs[64 * FLD];
  __shared__ __bf16 Ksw[8 * 512];   // [s*4+g][lane][j]
  __shared__ __bf16 Vsw[8 * 512];
  __shared__ __bf16 Ps[4][16 * FLD];

  const int tid = threadIdx.x, lane = tid & 31, w = tid >> 5;
  const int q0 = blockIdx.x * 64, ho = blockIdx.y * HD_;
  const int half = lane >> 4, nn = lane & 15;

  // stage Q (row-major)
  {
    const int r = tid >> 1, cb = (tid & 1) * 32;
    const __bf16* src = qb + (long)(q0 + r) * H_ + ho + cb;
#pragma unroll
    for (int jj = 0; jj < 4; ++jj)
      *(uint4*)&Qs[r * FLD + cb + jj * 8] = *(const uint4*)(src + jj * 8);
  }
  __syncthreads();

  const int qr = w * 16;
  v16bf qa[2];
  qa[0] = ld_frag_a(Qs, FLD, qr, 0, lane);
  qa[1] = ld_frag_a(Qs, FLD, qr, 32, lane);

  float runm[8], runs[8];
#pragma unroll
  for (int i = 0; i < 8; ++i) { runm[i] = -3.0e38f; runs[i] = 0.f; }
  v8f ofr[4] = {};

  const int nkt = blockIdx.x + 1;
  for (int kt = 0; kt < nkt; ++kt) {
    const int kv0 = kt * 64;
    __syncthreads();
    {
      // K: src kb[kv][hd] -> Ksw as B^T (n=kv, k=hd)
      const int kvr = tid >> 1, s = tid & 1;
      const __bf16* ks = kb + (long)(kv0 + kvr) * H_ + ho + s * 32;
#pragma unroll
      for (int jj = 0; jj < 4; ++jj) {
        const int hf = jj & 1, jb = (jj >> 1) * 8;
        __bf16* dst = &Ksw[(s * 4 + (kvr >> 4)) * 512 + (hf * 16 + (kvr & 15)) * 16 + jb];
        *(uint4*)dst = *(const uint4*)(ks + jj * 8);
      }
      // V: src vt[hd][t] -> Vsw as B (n=hd, k=kv)
      const int hd = tid >> 1, sc = tid & 1;
      const __bf16* vs = vt + (long)(ho + hd) * T_ + kv0 + sc * 32;
#pragma unroll
      for (int jj = 0; jj < 4; ++jj) {
        const int hf = jj & 1, jb = (jj >> 1) * 8;
        __bf16* dst = &Vsw[(sc * 4 + (hd >> 4)) * 512 + (hf * 16 + (hd & 15)) * 16 + jb];
        *(uint4*)dst = *(const uint4*)(vs + jj * 8);
      }
    }
    __syncthreads();

    // S = Q K^T
    v8f sf[4] = {};
#pragma unroll
    for (int s = 0; s < 2; ++s)
#pragma unroll
      for (int f = 0; f < 4; ++f) {
        v16bf kbf = ld_frag_contig(&Ksw[(s * 4 + f) * 512 + lane * 16]);
        sf[f] = wmma_bf16(qa[s], kbf, sf[f]);
      }

    // online softmax per row (row m = i + 8*half lives in one 16-lane half)
#pragma unroll
    for (int i = 0; i < 8; ++i) {
      const int gq = q0 + qr + i + 8 * half;
      float mloc = -3.0e38f;
#pragma unroll
      for (int f = 0; f < 4; ++f) {
        float sv = sf[f][i] * 0.125f;           // 1/sqrt(64)
        const int gk = kv0 + f * 16 + nn;
        if (gk > gq) sv = -3.0e38f;             // causal mask
        sf[f][i] = sv;
        mloc = fmaxf(mloc, sv);
      }
#pragma unroll
      for (int msk = 1; msk < 16; msk <<= 1) mloc = fmaxf(mloc, __shfl_xor(mloc, msk, 32));
      const float mnew = fmaxf(runm[i], mloc);
      const float corr = __expf(runm[i] - mnew);
      runm[i] = mnew;
      float rs = 0.f;
#pragma unroll
      for (int f = 0; f < 4; ++f) {
        float p = __expf(sf[f][i] - mnew);
        sf[f][i] = p;
        rs += p;
      }
#pragma unroll
      for (int msk = 1; msk < 16; msk <<= 1) rs += __shfl_xor(rs, msk, 32);
      runs[i] = runs[i] * corr + rs;
      ofr[0][i] *= corr; ofr[1][i] *= corr; ofr[2][i] *= corr; ofr[3][i] *= corr;
#pragma unroll
      for (int f = 0; f < 4; ++f)
        Ps[w][(i + 8 * half) * FLD + f * 16 + nn] = f2bf(sf[f][i]);
    }
    asm volatile("" ::: "memory");  // per-wave LDS order is HW-guaranteed; stop reordering

    // O += P V
#pragma unroll
    for (int s = 0; s < 2; ++s) {
      v16bf pa = ld_frag_a(Ps[w], FLD, 0, s * 32, lane);
#pragma unroll
      for (int f = 0; f < 4; ++f) {
        v16bf vbf = ld_frag_contig(&Vsw[(s * 4 + f) * 512 + lane * 16]);
        ofr[f] = wmma_bf16(pa, vbf, ofr[f]);
      }
    }
  }

  // epilogue: O /= l, store bf16 [T,H]
#pragma unroll
  for (int f = 0; f < 4; ++f)
#pragma unroll
    for (int i = 0; i < 8; ++i) {
      const int m = q0 + qr + i + 8 * half;
      o[(long)m * H_ + ho + f * 16 + nn] = f2bf(ofr[f][i] / runs[i]);
    }
}

// ---------------- SwiGLU elementwise ----------------
__global__ __launch_bounds__(256) void silu_mul_kernel(
    const float* __restrict__ g, const float* __restrict__ u, __bf16* __restrict__ a, long n) {
  long i = (long)blockIdx.x * 256 + threadIdx.x;
  if (i < n) {
    float gv = g[i];
    float s = gv / (1.f + __expf(-gv));
    a[i] = f2bf(s * u[i]);
  }
}

// ---------------- MoE: router + top2 ----------------
__global__ __launch_bounds__(256) void router_kernel(
    const __bf16* __restrict__ hn, const float* __restrict__ rw,
    float* __restrict__ probs, float* __restrict__ lse2,
    int* __restrict__ e0, int* __restrict__ e1,
    float* __restrict__ w0, float* __restrict__ w1) {
  const int t = blockIdx.x * 256 + threadIdx.x;
  if (t >= T_) return;
  float rl[E_];
#pragma unroll
  for (int e = 0; e < E_; ++e) rl[e] = 0.f;
  const __bf16* xr = hn + (long)t * H_;
  for (int h = 0; h < H_; ++h) {
    const float xv = bf2f(xr[h]);
    const float* wr = rw + h * E_;
#pragma unroll
    for (int e = 0; e < E_; ++e) rl[e] += xv * wr[e];
  }
  float mx = rl[0];
#pragma unroll
  for (int e = 1; e < E_; ++e) mx = fmaxf(mx, rl[e]);
  float se = 0.f, p[E_];
#pragma unroll
  for (int e = 0; e < E_; ++e) { p[e] = __expf(rl[e] - mx); se += p[e]; }
#pragma unroll
  for (int e = 0; e < E_; ++e) { p[e] /= se; probs[t * E_ + e] = p[e]; }
  const float lse = mx + __logf(se);
  lse2[t] = lse * lse;
  int b0 = 0;
#pragma unroll
  for (int e = 1; e < E_; ++e) if (p[e] > p[b0]) b0 = e;
  int b1 = (b0 == 0) ? 1 : 0;
#pragma unroll
  for (int e = 0; e < E_; ++e) if (e != b0 && p[e] > p[b1]) b1 = e;
  const float s2 = p[b0] + p[b1];
  e0[t] = b0; e1[t] = b1;
  w0[t] = p[b0] / s2; w1[t] = p[b1] / s2;
}

__global__ void moe_count_kernel(const int* __restrict__ e0, const int* __restrict__ e1,
                                 int* __restrict__ cnt) {
  const int e = threadIdx.x;
  if (e >= E_) return;
  int c = 0;
  for (int t = 0; t < T_; ++t) c += (e0[t] == e) + (e1[t] == e);
  cnt[e] = c;
}

__global__ void moe_offs_kernel(const int* __restrict__ cnt, int* __restrict__ aoff) {
  if (threadIdx.x == 0) {
    int o = 0;
    for (int e = 0; e < E_; ++e) { aoff[e] = o; o += (cnt[e] + 127) & ~127; }
    aoff[E_] = o;
  }
}

__global__ __launch_bounds__(256) void fill_init_kernel(int* __restrict__ assign) {
  const int i = blockIdx.x * 256 + threadIdx.x;
  if (i < CAP_) assign[i] = -1;
}

// deterministic per-expert serial fill (no atomics)
__global__ void moe_fill_kernel(const int* __restrict__ e0, const int* __restrict__ e1,
                                const int* __restrict__ aoff,
                                int* __restrict__ assign, int* __restrict__ ppos) {
  const int e = threadIdx.x;
  if (e >= E_) return;
  int pos = aoff[e];
  for (int t = 0; t < T_; ++t) {
    if (e0[t] == e)      { assign[pos] = t; ppos[2 * t]     = pos; ++pos; }
    else if (e1[t] == e) { assign[pos] = t; ppos[2 * t + 1] = pos; ++pos; }
  }
}

__global__ __launch_bounds__(256) void moe_combine_kernel(
    const float* __restrict__ dpack, const int* __restrict__ ppos,
    const float* __restrict__ w0, const float* __restrict__ w1, float* __restrict__ h) {
  long i = (long)blockIdx.x * 256 + threadIdx.x;  // T_*H_
  const int t = (int)(i >> 10), c = (int)(i & 1023);
  h[i] += w0[t] * dpack[(long)ppos[2 * t] * H_ + c] +
          w1[t] * dpack[(long)ppos[2 * t + 1] * H_ + c];
}

__global__ __launch_bounds__(256) void aux_kernel(
    const float* __restrict__ probs, const float* __restrict__ lse2,
    const int* __restrict__ cnt, float* __restrict__ outp) {
  const int tid = threadIdx.x;
  float pe[E_];
#pragma unroll
  for (int e = 0; e < E_; ++e) pe[e] = 0.f;
  float z = 0.f;
  for (int t = tid; t < T_; t += 256) {
    const float* pr = probs + (long)t * E_;
#pragma unroll
    for (int e = 0; e < E_; ++e) pe[e] += pr[e];
    z += lse2[t];
  }
  __shared__ float red[256];
  __shared__ float tot[E_];
  for (int e = 0; e < E_; ++e) {
    red[tid] = pe[e];
    __syncthreads();
    for (int s = 128; s > 0; s >>= 1) { if (tid < s) red[tid] += red[tid + s]; __syncthreads(); }
    if (tid == 0) tot[e] = red[0];
    __syncthreads();
  }
  red[tid] = z;
  __syncthreads();
  for (int s = 128; s > 0; s >>= 1) { if (tid < s) red[tid] += red[tid + s]; __syncthreads(); }
  if (tid == 0) {
    float aux = 0.f;
    for (int e = 0; e < E_; ++e)
      aux += ((float)cnt[e] / (float)T_) * (tot[e] / (float)T_);
    outp[0] = 0.01f * (float)E_ * aux + 0.001f * (red[0] / (float)T_);
  }
}

// ---------------- host orchestration ----------------
extern "C" void kernel_launch(void* const* d_in, const int* in_sizes, int n_in,
                              void* d_out, int out_size, void* d_ws, size_t ws_size,
                              hipStream_t stream) {
  (void)in_sizes; (void)n_in; (void)out_size; (void)ws_size;
  const int*   ids   = (const int*)  d_in[0];
  const float* embed = (const float*)d_in[1];
  const float* ln1   = (const float*)d_in[2];
  const float* ln2   = (const float*)d_in[3];
  const float* wq    = (const float*)d_in[4];
  const float* wk    = (const float*)d_in[5];
  const float* wv    = (const float*)d_in[6];
  const float* wo    = (const float*)d_in[7];
  const float* dgate = (const float*)d_in[8];
  const float* dup   = (const float*)d_in[9];
  const float* ddown = (const float*)d_in[10];
  const float* rw    = (const float*)d_in[11];
  const float* mgate = (const float*)d_in[12];
  const float* mup   = (const float*)d_in[13];
  const float* mdown = (const float*)d_in[14];
  const float* fnw   = (const float*)d_in[15];
  float* out = (float*)d_out;

  // bump allocator over d_ws
  char* wp = (char*)d_ws;
  auto alloc = [&](size_t bytes) -> void* {
    void* r = (void*)wp;
    wp += (bytes + 255) & ~(size_t)255;
    return r;
  };
  float*  hbuf  = (float*) alloc((size_t)T_ * H_ * 4);
  __bf16* hn    = (__bf16*)alloc((size_t)T_ * H_ * 2);
  float*  qf    = (float*) alloc((size_t)T_ * H_ * 4);
  float*  kf    = (float*) alloc((size_t)T_ * H_ * 4);
  float*  vf    = (float*) alloc((size_t)T_ * H_ * 4);
  __bf16* qb16  = (__bf16*)alloc((size_t)T_ * H_ * 2);
  __bf16* kb16  = (__bf16*)alloc((size_t)T_ * H_ * 2);
  __bf16* vtb   = (__bf16*)alloc((size_t)H_ * T_ * 2);
  __bf16* attno = (__bf16*)alloc((size_t)T_ * H_ * 2);
  float*  gbuf  = (float*) alloc((size_t)CAP_ * I_ * 4);
  float*  ubuf  = (float*) alloc((size_t)CAP_ * I_ * 4);
  __bf16* abuf  = (__bf16*)alloc((size_t)CAP_ * I_ * 2);
  float*  dpack = (float*) alloc((size_t)CAP_ * H_ * 4);
  float*  probs = (float*) alloc((size_t)T_ * E_ * 4);
  float*  lse2  = (float*) alloc((size_t)T_ * 4);
  int*    e0    = (int*)   alloc((size_t)T_ * 4);
  int*    e1    = (int*)   alloc((size_t)T_ * 4);
  float*  w0    = (float*) alloc((size_t)T_ * 4);
  float*  w1    = (float*) alloc((size_t)T_ * 4);
  int*    cnt   = (int*)   alloc(E_ * 4);
  int*    aoffb = (int*)   alloc((E_ + 1) * 4);
  int*    assign= (int*)   alloc((size_t)CAP_ * 4);
  int*    ppos  = (int*)   alloc((size_t)T_ * 2 * 4);
  // packed bf16 weights
  __bf16* p_wq  = (__bf16*)alloc((size_t)L_ * H_ * H_ * 2);
  __bf16* p_wk  = (__bf16*)alloc((size_t)L_ * H_ * H_ * 2);
  __bf16* p_wv  = (__bf16*)alloc((size_t)L_ * H_ * H_ * 2);
  __bf16* p_wo  = (__bf16*)alloc((size_t)L_ * H_ * H_ * 2);
  __bf16* p_dg  = (__bf16*)alloc((size_t)3 * H_ * I_ * 2);
  __bf16* p_du  = (__bf16*)alloc((size_t)3 * H_ * I_ * 2);
  __bf16* p_dd  = (__bf16*)alloc((size_t)3 * I_ * H_ * 2);
  __bf16* p_mg  = (__bf16*)alloc((size_t)E_ * H_ * I_ * 2);
  __bf16* p_mu  = (__bf16*)alloc((size_t)E_ * H_ * I_ * 2);
  __bf16* p_md  = (__bf16*)alloc((size_t)E_ * I_ * H_ * 2);
  __bf16* p_emb = (__bf16*)alloc((size_t)H_ * V_ * 2);

  auto pack = [&](const float* s, __bf16* d, int Kd, int N, int tB, int nmat) {
    long total = (long)nmat * Kd * N;
    pack_w_kernel<<<(int)((total + 255) / 256), 256, 0, stream>>>(s, d, Kd, N, tB, nmat);
  };
  pack(wq, p_wq, H_, H_, 0, L_);
  pack(wk, p_wk, H_, H_, 0, L_);
  pack(wv, p_wv, H_, H_, 0, L_);
  pack(wo, p_wo, H_, H_, 0, L_);
  pack(dgate, p_dg, H_, I_, 0, 3);
  pack(dup,   p_du, H_, I_, 0, 3);
  pack(ddown, p_dd, I_, H_, 0, 3);
  pack(mgate, p_mg, H_, I_, 0, E_);
  pack(mup,   p_mu, H_, I_, 0, E_);
  pack(mdown, p_md, I_, H_, 0, E_);
  pack(embed, p_emb, H_, V_, 1, 1);

  auto gemm = [&](const __bf16* A, int lda, const __bf16* Bp, long sB,
                  float* C, int ldc, int M, int N, int Kd,
                  const float* res, const int* ridx, const int* aoff_, int gridM) {
    dim3 g(gridM, N / GBN);
    gemm_bf16_kernel<<<g, 256, 0, stream>>>(A, lda, Bp, sB, N / GBN, C, ldc,
                                            M, N, Kd, res, ridx, aoff_);
  };

  gather_embed_kernel<<<(T_ * H_) / 256, 256, 0, stream>>>(ids, embed, hbuf);

  for (int li = 0; li < L_; ++li) {
    const long oHH = (long)li * H_ * H_;
    // ---- attention ----
    rmsnorm_kernel<<<T_, 256, 0, stream>>>(hbuf, ln1 + (size_t)li * H_, hn);
    gemm(hn, H_, p_wq + oHH, 0, qf, H_, T_, H_, H_, nullptr, nullptr, nullptr, T_ / GBM);
    gemm(hn, H_, p_wk + oHH, 0, kf, H_, T_, H_, H_, nullptr, nullptr, nullptr, T_ / GBM);
    gemm(hn, H_, p_wv + oHH, 0, vf, H_, T_, H_, H_, nullptr, nullptr, nullptr, T_ / GBM);
    rope_bf16_kernel<<<(T_ * NH_ * 32) / 256, 256, 0, stream>>>(qf, qb16);
    rope_bf16_kernel<<<(T_ * NH_ * 32) / 256, 256, 0, stream>>>(kf, kb16);
    vtrans_kernel<<<(T_ * H_) / 256, 256, 0, stream>>>(vf, vtb);
    flash_attn_kernel<<<dim3(T_ / 64, NH_), 128, 0, stream>>>(qb16, kb16, vtb, attno);
    gemm(attno, H_, p_wo + oHH, 0, hbuf, H_, T_, H_, H_, hbuf, nullptr, nullptr, T_ / GBM);

    // ---- FFN ----
    rmsnorm_kernel<<<T_, 256, 0, stream>>>(hbuf, ln2 + (size_t)li * H_, hn);
    if (li == 2) {
      router_kernel<<<T_ / 256, 256, 0, stream>>>(hn, rw, probs, lse2, e0, e1, w0, w1);
      moe_count_kernel<<<1, 32, 0, stream>>>(e0, e1, cnt);
      moe_offs_kernel<<<1, 32, 0, stream>>>(cnt, aoffb);
      fill_init_kernel<<<(CAP_ + 255) / 256, 256, 0, stream>>>(assign);
      moe_fill_kernel<<<1, 32, 0, stream>>>(e0, e1, aoffb, assign, ppos);
      gemm(hn, H_, p_mg, (long)H_ * I_, gbuf, I_, CAP_, I_, H_, nullptr, assign, aoffb, CAP_ / GBM);
      gemm(hn, H_, p_mu, (long)H_ * I_, ubuf, I_, CAP_, I_, H_, nullptr, assign, aoffb, CAP_ / GBM);
      silu_mul_kernel<<<(int)(((long)CAP_ * I_ + 255) / 256), 256, 0, stream>>>(gbuf, ubuf, abuf, (long)CAP_ * I_);
      gemm(abuf, I_, p_md, (long)I_ * H_, dpack, H_, CAP_, H_, I_, nullptr, nullptr, aoffb, CAP_ / GBM);
      moe_combine_kernel<<<(T_ * H_) / 256, 256, 0, stream>>>(dpack, ppos, w0, w1, hbuf);
      aux_kernel<<<1, 256, 0, stream>>>(probs, lse2, cnt, out + (size_t)T_ * V_);
    } else {
      const int d = (li < 2) ? li : li - 1;
      gemm(hn, H_, p_dg + (size_t)d * H_ * I_, 0, gbuf, I_, T_, I_, H_, nullptr, nullptr, nullptr, T_ / GBM);
      gemm(hn, H_, p_du + (size_t)d * H_ * I_, 0, ubuf, I_, T_, I_, H_, nullptr, nullptr, nullptr, T_ / GBM);
      silu_mul_kernel<<<(int)(((long)T_ * I_ + 255) / 256), 256, 0, stream>>>(gbuf, ubuf, abuf, (long)T_ * I_);
      gemm(abuf, I_, p_dd + (size_t)d * I_ * H_, 0, hbuf, H_, T_, H_, I_, hbuf, nullptr, nullptr, T_ / GBM);
    }
  }

  // ---- final norm + logits (packed embed^T) ----
  rmsnorm_kernel<<<T_, 256, 0, stream>>>(hbuf, fnw, hn);
  gemm(hn, H_, p_emb, 0, out, V_, T_, V_, H_, nullptr, nullptr, nullptr, T_ / GBM);
}
